// ALLSTMCell_19791209300394
// MI455X (gfx1250) — matrix operand order repaired and checked by reading, not compiled, not run on hardware
//
#include <hip/hip_runtime.h>

// ---------------------------------------------------------------------------
// LSTM cell for MI455X (gfx1250, wave32, WMMA + async global->LDS DMA):
//   gates = W_ih @ x + W_hh @ h + b   (M=16384, K=4096+4096, N=1024)
//   out   = sum(c_next + h_next)      (scalar)
// Weights stay fp32 in HBM (read exactly once, ~512MB ~= 22us @ 23.3TB/s),
// async-copied raw into LDS, converted to f16 at fragment load;
// activations pre-transposed to f16 (L2-resident, 16MB).
// ---------------------------------------------------------------------------

typedef __attribute__((ext_vector_type(16))) _Float16 v16h;
typedef __attribute__((ext_vector_type(8)))  float    v8f;

#define D_IN   4096
#define H_HID  4096
#define B_COL  1024
#define BM     128        // 4 gates x 32 hidden rows
#define BN     128
#define BK     32
#define LDA    36         // A tile row stride in floats  (32 + 4 pad -> 144B)
#define LDB    40         // B tile row stride in halfs   (32 + 8 pad -> 80B)
#define NKT    256        // (D_IN + H_HID) / BK
#define LDC    68         // epilogue staging stride (floats)

union __align__(16) SMem {
  struct {
    float          a[2][BM * LDA];    // fp32 weight tiles   2*18432B
    unsigned short b[2][BM * LDB];    // f16 activation tiles 2*10240B
  } t;                                // total 57344B
  float cst[BM * LDC];                // 128 x 64 f32 gate staging 34816B
};

__device__ __forceinline__ float sigmoidf_(float x) {
  return 1.0f / (1.0f + __expf(-x));
}

// async 16B global -> LDS copy (per lane), tracked by ASYNCcnt
__device__ __forceinline__ void async_copy_b128(unsigned lds_off, const void* g) {
  asm volatile("global_load_async_to_lds_b128 %0, %1, off"
               :
               : "v"(lds_off), "v"((unsigned long long)(uintptr_t)g)
               : "memory");
}
__device__ __forceinline__ void wait_async0() {
  asm volatile("s_wait_asynccnt 0x0" ::: "memory");
}
__device__ __forceinline__ unsigned lds_addr_of(const void* p) {
  // LDS aperture: generic address low 32 bits == LDS byte address
  return (unsigned)(uintptr_t)p;
}

// --------- transpose + convert: out[n][k] = (f16) in[k][n], in is R x C ----
__global__ __launch_bounds__(256)
void cvtT_kernel(const float* __restrict__ in, _Float16* __restrict__ out,
                 int R, int C) {
  __shared__ float tile[32][33];
  const int tx = threadIdx.x, ty = threadIdx.y;
  const int c = blockIdx.x * 32 + tx;
#pragma unroll
  for (int i = 0; i < 32; i += 8) {
    int r = blockIdx.y * 32 + ty + i;
    tile[ty + i][tx] = in[(size_t)r * C + c];
  }
  __syncthreads();
  const int k = blockIdx.y * 32 + tx;
#pragma unroll
  for (int i = 0; i < 32; i += 8) {
    int n = blockIdx.x * 32 + ty + i;
    out[(size_t)n * R + k] = (_Float16)tile[tx][ty + i];
  }
}

// --------------------------- fused GEMM + LSTM -----------------------------
__global__ __launch_bounds__(256)
void lstm_gemm_kernel(const float* __restrict__ wih,   // (4H, D) fp32
                      const float* __restrict__ whh,   // (4H, H) fp32
                      const _Float16* __restrict__ xt, // (B, D)  f16 (x^T)
                      const _Float16* __restrict__ ht, // (B, H)  f16 (h^T)
                      const float* __restrict__ c_prev,// (H, B)  fp32
                      const float* __restrict__ bih,   // (4H)
                      const float* __restrict__ bhh,   // (4H)
                      float* __restrict__ partials) {
  __shared__ SMem smem;
  __shared__ float red[8];

  const int tid  = threadIdx.x;
  const int lane = tid & 31;
  const int wave = tid >> 5;
  const int wm   = wave >> 1;            // 0..3  -> gate block (32 M-rows)
  const int wn   = wave & 1;             // 0..1  -> 64-col block
  const int m0   = blockIdx.y * 32;      // hidden-row block
  const int n0   = blockIdx.x * BN;      // batch-col block

  // ---- per-thread async staging descriptors -------------------------------
  // A tile: 128 rows x 32 f32 = 1024 x 16B chunks; thread handles 4 chunks,
  //         one per gate block j: row_local = tid>>3 + 32*j (gate j).
  const int arl  = tid >> 3;             // 0..31 (row within gate block)
  const int akf  = (tid & 7) * 4;        // float offset within row
  // B tile: 128 rows x 32 f16 = 512 x 16B chunks; thread handles 2 chunks.
  const int brl  = tid >> 2;             // 0..63
  const int bk8  = (tid & 3) * 8;        // half offset within row

  size_t aoffe[4], boffe[2];
#pragma unroll
  for (int j = 0; j < 4; ++j)
    aoffe[j] = (size_t)(j * H_HID + m0 + arl) * D_IN + akf;
#pragma unroll
  for (int j = 0; j < 2; ++j)
    boffe[j] = (size_t)(n0 + brl + 64 * j) * D_IN + bk8;

  unsigned aLds[2][4], bLds[2][2];
#pragma unroll
  for (int bu = 0; bu < 2; ++bu) {
    unsigned ab = lds_addr_of(&smem.t.a[bu][0]);
    unsigned bb = lds_addr_of(&smem.t.b[bu][0]);
#pragma unroll
    for (int j = 0; j < 4; ++j)
      aLds[bu][j] = ab + ((arl + 32 * j) * LDA + akf) * 4u;
#pragma unroll
    for (int j = 0; j < 2; ++j)
      bLds[bu][j] = bb + ((brl + 64 * j) * LDB + bk8) * 2u;
  }

  auto issue_async = [&](int kt, int bu) {
    int kk = kt * BK;
    const float*    w = wih;
    const _Float16* a = xt;
    if (kk >= D_IN) { w = whh; a = ht; kk -= D_IN; }
#pragma unroll
    for (int j = 0; j < 4; ++j)
      async_copy_b128(aLds[bu][j], w + aoffe[j] + kk);
#pragma unroll
    for (int j = 0; j < 2; ++j)
      async_copy_b128(bLds[bu][j], a + boffe[j] + kk);
  };

  v8f acc[2][4];
#pragma unroll
  for (int i = 0; i < 2; ++i)
#pragma unroll
    for (int j = 0; j < 4; ++j) acc[i][j] = {};

  const int lr = lane & 15;
  const int ah = (lane >> 4) * 8;   // A frag f32 offsets: ah..ah+7, ah+16..ah+23
  const int bh = (lane >> 4) * 8;   // B frag: halfs k=bh..bh+7, bh+16..bh+23

  issue_async(0, 0);
  wait_async0();
  __syncthreads();

  for (int kt = 0; kt < NKT; ++kt) {
    const int buf = kt & 1;
    if (kt + 1 < NKT) issue_async(kt + 1, buf ^ 1);

    const float*          smA = smem.t.a[buf];
    const unsigned short* smB = smem.t.b[buf];

    union { _Float16 h[16]; v16h v; } af[2];
    union { uint4 u[2]; v16h v; }    bf[4];
#pragma unroll
    for (int mi = 0; mi < 2; ++mi) {
      const float* p = smA + (wm * 32 + mi * 16 + lr) * LDA;
      float4 f0 = *reinterpret_cast<const float4*>(p + ah);
      float4 f1 = *reinterpret_cast<const float4*>(p + ah + 4);
      float4 f2 = *reinterpret_cast<const float4*>(p + ah + 16);
      float4 f3 = *reinterpret_cast<const float4*>(p + ah + 20);
      af[mi].h[0]  = (_Float16)f0.x; af[mi].h[1]  = (_Float16)f0.y;
      af[mi].h[2]  = (_Float16)f0.z; af[mi].h[3]  = (_Float16)f0.w;
      af[mi].h[4]  = (_Float16)f1.x; af[mi].h[5]  = (_Float16)f1.y;
      af[mi].h[6]  = (_Float16)f1.z; af[mi].h[7]  = (_Float16)f1.w;
      af[mi].h[8]  = (_Float16)f2.x; af[mi].h[9]  = (_Float16)f2.y;
      af[mi].h[10] = (_Float16)f2.z; af[mi].h[11] = (_Float16)f2.w;
      af[mi].h[12] = (_Float16)f3.x; af[mi].h[13] = (_Float16)f3.y;
      af[mi].h[14] = (_Float16)f3.z; af[mi].h[15] = (_Float16)f3.w;
    }
#pragma unroll
    for (int ni = 0; ni < 4; ++ni) {
      const unsigned short* p = smB + (wn * 64 + ni * 16 + lr) * LDB + bh;
      bf[ni].u[0] = *reinterpret_cast<const uint4*>(p);
      bf[ni].u[1] = *reinterpret_cast<const uint4*>(p + 16);
    }
#pragma unroll
    for (int mi = 0; mi < 2; ++mi)
#pragma unroll
      for (int ni = 0; ni < 4; ++ni)
        acc[mi][ni] = __builtin_amdgcn_wmma_f32_16x16x32_f16(
            false, af[mi].v, false, bf[ni].v, (short)0, acc[mi][ni],
            false, false);

    wait_async0();
    __syncthreads();
  }

  // ---------------- epilogue: gates -> c_next/h_next -> sum ----------------
  float lsum = 0.0f;
#pragma unroll 1
  for (int ph = 0; ph < 2; ++ph) {
    if (wn == ph) {
#pragma unroll
      for (int mi = 0; mi < 2; ++mi)
#pragma unroll
        for (int ni = 0; ni < 4; ++ni)
#pragma unroll
          for (int j = 0; j < 8; ++j) {
            int M = wm * 32 + mi * 16 + (lane >> 4) * 8 + j; // tile row
            int N = ni * 16 + lr;                            // col within 64
            smem.cst[M * LDC + N] = acc[mi][ni][j];
          }
    }
    __syncthreads();
    for (int e = tid; e < 32 * 64; e += 256) {
      int rr   = e >> 6;
      int cc   = e & 63;
      int hrow = m0 + rr;
      int col  = n0 + ph * 64 + cc;
      float iv = smem.cst[(rr      ) * LDC + cc] + bih[hrow            ] + bhh[hrow            ];
      float fv = smem.cst[(rr + 32) * LDC + cc] + bih[hrow +     H_HID] + bhh[hrow +     H_HID];
      float gv = smem.cst[(rr + 64) * LDC + cc] + bih[hrow + 2 * H_HID] + bhh[hrow + 2 * H_HID];
      float ov = smem.cst[(rr + 96) * LDC + cc] + bih[hrow + 3 * H_HID] + bhh[hrow + 3 * H_HID];
      float ig = sigmoidf_(iv);
      float fg = sigmoidf_(fv);
      float gg = tanhf(gv);
      float og = sigmoidf_(ov);
      float cp = c_prev[(size_t)hrow * B_COL + col];
      float cn = fg * cp + ig * gg;
      float hn = og * tanhf(cn);
      lsum += cn + hn;
    }
    __syncthreads();
  }

  // deterministic block reduction
#pragma unroll
  for (int off = 16; off > 0; off >>= 1)
    lsum += __shfl_down(lsum, off, 32);
  if (lane == 0) red[wave] = lsum;
  __syncthreads();
  if (tid == 0) {
    float s = 0.0f;
#pragma unroll
    for (int wv = 0; wv < 8; ++wv) s += red[wv];
    partials[blockIdx.y * gridDim.x + blockIdx.x] = s;
  }
}

// --------------------------- final reduction -------------------------------
__global__ __launch_bounds__(256)
void final_reduce_kernel(const float* __restrict__ partials,
                         float* __restrict__ out, int n) {
  __shared__ float red[256];
  float s = 0.0f;
  for (int i = threadIdx.x; i < n; i += 256) s += partials[i];
  red[threadIdx.x] = s;
  __syncthreads();
  for (int off = 128; off > 0; off >>= 1) {
    if (threadIdx.x < off) red[threadIdx.x] += red[threadIdx.x + off];
    __syncthreads();
  }
  if (threadIdx.x == 0) out[0] = red[0];
}

// ---------------------------------------------------------------------------
extern "C" void kernel_launch(void* const* d_in, const int* in_sizes, int n_in,
                              void* d_out, int out_size, void* d_ws, size_t ws_size,
                              hipStream_t stream) {
  (void)in_sizes; (void)n_in; (void)out_size; (void)ws_size;
  const float* x   = (const float*)d_in[0];
  const float* h   = (const float*)d_in[1];
  const float* c   = (const float*)d_in[2];
  const float* wih = (const float*)d_in[3];
  const float* whh = (const float*)d_in[4];
  const float* bih = (const float*)d_in[5];
  const float* bhh = (const float*)d_in[6];
  float* out = (float*)d_out;

  const size_t ACT_ELEMS = (size_t)D_IN * B_COL; // 4 Mi elements
  char* ws = (char*)d_ws;
  _Float16* xt16 = (_Float16*)ws; ws += ACT_ELEMS * sizeof(_Float16);
  _Float16* ht16 = (_Float16*)ws; ws += ACT_ELEMS * sizeof(_Float16);
  float* partials = (float*)ws;

  dim3 tb(32, 8);
  cvtT_kernel<<<dim3(B_COL / 32, D_IN / 32), tb, 0, stream>>>(x, xt16, D_IN, B_COL);
  cvtT_kernel<<<dim3(B_COL / 32, H_HID / 32), tb, 0, stream>>>(h, ht16, H_HID, B_COL);

  lstm_gemm_kernel<<<dim3(B_COL / BN, H_HID / 32), 256, 0, stream>>>(
      wih, whh, xt16, ht16, c, bih, bhh, partials);

  final_reduce_kernel<<<1, 256, 0, stream>>>(
      partials, out, (B_COL / BN) * (H_HID / 32));
}
